// SelfAttention_74345883894460
// MI455X (gfx1250) — compile-verified, matrix operand order checked by reading, not run on hardware
//
#include <hip/hip_runtime.h>

// ---------------------------------------------------------------------------
// CDNA5 (gfx1250) wave32 WMMA channel self-attention.
// All LDS tiles are stored in WMMA *fragment layout*: per 16x32 (A) or 32x16
// (B) tile, lane L owns 16 contiguous bf16 (32 B) -> fragment load = 2x
// ds_load_b128, no lane shuffles. Producers scatter into this layout.
// ---------------------------------------------------------------------------

typedef __attribute__((ext_vector_type(16))) __bf16 v16bf;
typedef __attribute__((ext_vector_type(8)))  float  v8f;

constexpr int Bn  = 8;       // batch
constexpr int Cc  = 512;     // channels
constexpr int ICc = 64;      // inner channels
constexpr int Nn  = 16384;   // spatial (128*128)
constexpr int NT  = 128;     // spatial tile per block
constexpr int KC  = 32;      // WMMA K per step
constexpr int FR  = 512;     // bf16 elements per fragment tile (32 lanes x 16)

__device__ __forceinline__ unsigned short bfbits(float f) {
  unsigned u = __builtin_bit_cast(unsigned, f);
  unsigned r = u + 0x7FFFu + ((u >> 16) & 1u);   // round-to-nearest-even
  return (unsigned short)(r >> 16);
}
__device__ __forceinline__ __bf16 f2bf(float f) {
  unsigned short h = bfbits(f);
  return __builtin_bit_cast(__bf16, h);
}

// ISA 7.12.2 16-bit A/B fragment maps: K index -> (lane half, element slot).
__device__ __forceinline__ int emap(int k) {     // element index within lane
  return 2 * (((k & 7) >> 1) + ((k >> 4) << 2)) + (k & 1);
}
__device__ __forceinline__ int lhalf(int k) { return (k >> 3) & 1; }

// Fragment tile load from LDS (tile stored as [lane][16 bf16], 32B/lane).
__device__ __forceinline__ v16bf load_frag_lds(const __bf16* tile) {
  return *((const v16bf*)tile + (threadIdx.x & 31));
}

// A fragment from row-major global matrix: element (m,k) = p[m*ld + k].
// (Pairs are contiguous -> compiler merges to 2x global_load_b128.)
__device__ __forceinline__ v16bf load_frag_row(const __bf16* p, int ld) {
  int lane = threadIdx.x & 31;
  const __bf16* row = p + (lane & 15) * ld;
  int kg = (lane >> 4) << 3;
  v16bf f;
#pragma unroll
  for (int r = 0; r < 8; ++r) {
    int k = (r < 4 ? 2 * r : 16 + 2 * (r - 4)) + kg;
    f[2 * r]     = row[k];
    f[2 * r + 1] = row[k + 1];
  }
  return f;
}

__device__ __forceinline__ v8f wmma_bf16(v16bf a, v16bf b, v8f c) {
  return __builtin_amdgcn_wmma_f32_16x16x32_bf16(false, a, false, b,
                                                 (short)0, c, false, false);
}

// Pack 4 bf16 and store as one 8-byte LDS write (dst is 8B aligned).
__device__ __forceinline__ void store4f(__bf16* dst, float a, float b,
                                        float c, float d) {
  unsigned long long p = (unsigned long long)bfbits(a)
                       | ((unsigned long long)bfbits(b) << 16)
                       | ((unsigned long long)bfbits(c) << 32)
                       | ((unsigned long long)bfbits(d) << 48);
  *(unsigned long long*)dst = p;
}
__device__ __forceinline__ void store4h(__bf16* dst, __bf16 a, __bf16 b,
                                        __bf16 c, __bf16 d) {
  unsigned long long p = (unsigned long long)__builtin_bit_cast(unsigned short, a)
                       | ((unsigned long long)__builtin_bit_cast(unsigned short, b) << 16)
                       | ((unsigned long long)__builtin_bit_cast(unsigned short, c) << 32)
                       | ((unsigned long long)__builtin_bit_cast(unsigned short, d) << 48);
  *(unsigned long long*)dst = p;
}

// ---------------------------------------------------------------------------
// Kernel 0: weights -> bf16, zero the score accumulator.
// ---------------------------------------------------------------------------
__global__ void k_prep(const float* __restrict__ Wq, const float* __restrict__ Wk,
                       const float* __restrict__ Wv, const float* __restrict__ Wr,
                       __bf16* __restrict__ wqb, __bf16* __restrict__ wkb,
                       __bf16* __restrict__ wvb, __bf16* __restrict__ wrb,
                       float* __restrict__ scores) {
  int i = blockIdx.x * blockDim.x + threadIdx.x;
  if (i < ICc * Cc) {
    wqb[i] = f2bf(Wq[i]);
    wkb[i] = f2bf(Wk[i]);
    wvb[i] = f2bf(Wv[i]);
    wrb[i] = f2bf(Wr[i]);
    scores[i] = 0.0f;   // 8*64*64 == 64*512
  }
}

// ---------------------------------------------------------------------------
// Kernel 1: fused QKV + partial score accumulation.
// Grid: B * (N/NT), 256 threads (8 waves); wave w owns 16 columns.
// ---------------------------------------------------------------------------
__global__ __launch_bounds__(256) void k_qkv_scores(
    const float* __restrict__ x,
    const __bf16* __restrict__ wq, const __bf16* __restrict__ wk,
    const __bf16* __restrict__ wv,
    const float* __restrict__ bq, const float* __restrict__ bk,
    const float* __restrict__ bv,
    __bf16* __restrict__ vws, float* __restrict__ scores) {
  // x chunk: 8 col-tiles (B frags);  Q/K: [rowtile(4) x kchunk(4)] frag tiles.
  __shared__ __attribute__((aligned(32))) __bf16 xfrag[8 * FR];    //  8 KB
  __shared__ __attribute__((aligned(32))) __bf16 qfrag[16 * FR];   // 16 KB
  __shared__ __attribute__((aligned(32))) __bf16 kfrag[16 * FR];   // 16 KB

  const int b    = blockIdx.x >> 7;
  const int n0   = (blockIdx.x & 127) * NT;
  const int tid  = threadIdx.x;
  const int wave = tid >> 5;
  const int lane = tid & 31;

  // staging coordinates: column c, 4-row group start kq (0 or 4).
  const int c  = tid & 127;
  const int kq = (tid >> 7) << 2;
  const int ct = c >> 4;

  v8f aq[4], ak[4], av[4];
#pragma unroll
  for (int rt = 0; rt < 4; ++rt) { aq[rt] = {}; ak[rt] = {}; av[rt] = {}; }

  const float* xb = x + (size_t)b * Cc * Nn + n0;

  for (int kc = 0; kc < Cc / KC; ++kc) {
    __syncthreads();                       // previous chunk fully consumed
#pragma unroll
    for (int j = 0; j < 4; ++j) {
      int kb = kq + 8 * j;                 // 4-aligned row group within chunk
      const float* src = xb + (size_t)(kc * KC + kb) * Nn + c;
      __bf16* dst = xfrag + ct * FR
                  + (lhalf(kb) * 16 + (c & 15)) * 16 + emap(kb);
      store4f(dst, src[0], src[Nn], src[2 * Nn], src[3 * Nn]);
    }
    __syncthreads();

    v16bf bf = load_frag_lds(xfrag + wave * FR);  // 2x ds_load_b128, 12 reuses
#pragma unroll
    for (int rt = 0; rt < 4; ++rt) {
      const int woff = rt * 16 * Cc + kc * KC;
      aq[rt] = wmma_bf16(load_frag_row(wq + woff, Cc), bf, aq[rt]);
      ak[rt] = wmma_bf16(load_frag_row(wk + woff, Cc), bf, ak[rt]);
      av[rt] = wmma_bf16(load_frag_row(wv + woff, Cc), bf, av[rt]);
    }
  }

  // Bias add; V -> workspace (bf16); Q/K -> LDS in fragment layout for the
  // score matmul (Q as A-frag, K as B^T-frag share the same addressing).
  {
    const int n    = lane & 15;
    const int mh   = (lane >> 4) << 3;     // C/D: m = vgpr + 8*(lane/16)
    const int sidx = wave >> 1;            // spatial K-chunk (0..3)
    const int kcol = ((wave & 1) << 4) + n;  // column within 32-chunk
    const int lt   = lhalf(kcol) << 4;
    const int ec   = emap(kcol);
#pragma unroll
    for (int rt = 0; rt < 4; ++rt) {
#pragma unroll
      for (int v = 0; v < 8; ++v) {
        int m   = mh + v;
        int row = rt * 16 + m;
        int fo  = (rt * 4 + sidx) * FR + (lt + m) * 16 + ec;
        qfrag[fo] = f2bf(aq[rt][v] + bq[row]);
        kfrag[fo] = f2bf(ak[rt][v] + bk[row]);
        vws[((size_t)b * ICc + row) * Nn + n0 + wave * 16 + n] =
            f2bf(av[rt][v] + bv[row]);
      }
    }
  }
  __syncthreads();

  // Partial scores S[i][j] += sum_n Q[i,n]K[j,n]; 16 tiles, 2 per wave
  // (both tiles share the same i row-tile -> shared A fragment).
  const int it  = (wave * 2) >> 2;
  const int jt0 = (wave * 2) & 3;
  v8f sc[2] = {{}, {}};
#pragma unroll
  for (int s = 0; s < NT / KC; ++s) {
    v16bf a = load_frag_lds(qfrag + (it * 4 + s) * FR);
    sc[0] = wmma_bf16(a, load_frag_lds(kfrag + (jt0 * 4 + s) * FR), sc[0]);
    sc[1] = wmma_bf16(a, load_frag_lds(kfrag + ((jt0 + 1) * 4 + s) * FR), sc[1]);
  }
  {
    const int n  = lane & 15;
    const int mh = (lane >> 4) << 3;
    float* sb = scores + (size_t)b * ICc * ICc;
#pragma unroll
    for (int t = 0; t < 2; ++t) {
      int jt = jt0 + t;
#pragma unroll
      for (int v = 0; v < 8; ++v) {
        int i = it * 16 + mh + v;
        int j = jt * 16 + n;
        atomicAdd(sb + i * ICc + j, sc[t][v]);
      }
    }
  }
}

// ---------------------------------------------------------------------------
// Kernel 2: softmax over axis i (dim=1) for each (b, j). Tiny.
// ---------------------------------------------------------------------------
__global__ void k_softmax(const float* __restrict__ scores,
                          __bf16* __restrict__ wgt) {
  int b = blockIdx.x;
  int j = threadIdx.x;                    // 0..63
  const float* s = scores + b * ICc * ICc;
  float m = -3.402823466e38f;
  for (int i = 0; i < ICc; ++i) m = fmaxf(m, s[i * ICc + j]);
  float sum = 0.0f;
  for (int i = 0; i < ICc; ++i) sum += __expf(s[i * ICc + j] - m);
  float inv = 1.0f / sum;
  for (int i = 0; i < ICc; ++i)
    wgt[b * ICc * ICc + i * ICc + j] = f2bf(__expf(s[i * ICc + j] - m) * inv);
}

// ---------------------------------------------------------------------------
// Kernel 3: att = softmax(S)*V, out = Wr*att + br + x (residual fused).
// ---------------------------------------------------------------------------
__global__ __launch_bounds__(256) void k_att_out(
    const float* __restrict__ x, const __bf16* __restrict__ wgt,
    const __bf16* __restrict__ vws, const __bf16* __restrict__ wr,
    const float* __restrict__ br, float* __restrict__ out) {
  // V and att tiles as B fragments: [kchunk(2)][coltile(8)].
  __shared__ __attribute__((aligned(32))) __bf16 vfrag[16 * FR];   // 16 KB
  __shared__ __attribute__((aligned(32))) __bf16 afrag[16 * FR];   // 16 KB

  const int b    = blockIdx.x >> 7;
  const int n0   = (blockIdx.x & 127) * NT;
  const int tid  = threadIdx.x;
  const int wave = tid >> 5;
  const int lane = tid & 31;

  // Stage V tile [64 x 128] into fragment layout (1x ds_store_b64 / group).
  {
    const int c  = tid & 127;
    const int kq = (tid >> 7) << 2;
    const int ct = c >> 4;
    const __bf16* vsrc = vws + (size_t)b * ICc * Nn + n0 + c;
#pragma unroll
    for (int j = 0; j < 8; ++j) {
      int kb = kq + 8 * j;                 // 0..60, 4-aligned
      int s  = kb >> 5, k = kb & 31;
      const __bf16* p = vsrc + (size_t)kb * Nn;
      __bf16* dst = vfrag + (s * 8 + ct) * FR
                  + (lhalf(k) * 16 + (c & 15)) * 16 + emap(k);
      store4h(dst, p[0], p[Nn], p[2 * Nn], p[3 * Nn]);
    }
  }
  __syncthreads();

  // att tile [64 x 128]: 4 row-tiles per wave, K = 64 (2 steps).
  const __bf16* wb = wgt + b * ICc * ICc;
  v8f at[4];
#pragma unroll
  for (int rt = 0; rt < 4; ++rt) at[rt] = {};
#pragma unroll
  for (int s = 0; s < 2; ++s) {
    v16bf bf = load_frag_lds(vfrag + (s * 8 + wave) * FR);
#pragma unroll
    for (int rt = 0; rt < 4; ++rt)
      at[rt] = wmma_bf16(load_frag_row(wb + rt * 16 * ICc + s * KC, ICc),
                         bf, at[rt]);
  }
  // att -> LDS as B fragments for the output projection (per-wave column).
  {
    const int n  = lane & 15;
    const int mh = (lane >> 4) << 3;
#pragma unroll
    for (int rt = 0; rt < 4; ++rt)
#pragma unroll
      for (int v = 0; v < 8; ++v) {
        int row = rt * 16 + mh + v;        // K index of out matmul
        int s = row >> 5, k = row & 31;
        afrag[(s * 8 + wave) * FR + (lhalf(k) * 16 + n) * 16 + emap(k)] =
            f2bf(at[rt][v]);
      }
  }
  __syncthreads();

  // out tile [512 x 128]: per wave 32 row-tiles; B fragments hoisted.
  const float* xb = x + (size_t)b * Cc * Nn + n0;
  float* ob = out + (size_t)b * Cc * Nn + n0;
  const int n  = lane & 15;
  const int mh = (lane >> 4) << 3;
  v16bf bf0 = load_frag_lds(afrag + (0 * 8 + wave) * FR);
  v16bf bf1 = load_frag_lds(afrag + (1 * 8 + wave) * FR);
  for (int rt = 0; rt < 32; ++rt) {
    v8f acc = {};
    acc = wmma_bf16(load_frag_row(wr + rt * 16 * ICc, ICc),      bf0, acc);
    acc = wmma_bf16(load_frag_row(wr + rt * 16 * ICc + KC, ICc), bf1, acc);
#pragma unroll
    for (int v = 0; v < 8; ++v) {
      int row = rt * 16 + mh + v;
      size_t idx = (size_t)row * Nn + wave * 16 + n;
      ob[idx] = acc[v] + br[row] + xb[idx];
    }
  }
}

// ---------------------------------------------------------------------------
// Host entry
// ---------------------------------------------------------------------------
extern "C" void kernel_launch(void* const* d_in, const int* in_sizes, int n_in,
                              void* d_out, int out_size, void* d_ws, size_t ws_size,
                              hipStream_t stream) {
  (void)in_sizes; (void)n_in; (void)out_size; (void)ws_size;
  const float* x  = (const float*)d_in[0];
  const float* Wq = (const float*)d_in[1];
  const float* bq = (const float*)d_in[2];
  const float* Wk = (const float*)d_in[3];
  const float* bk = (const float*)d_in[4];
  const float* Wv = (const float*)d_in[5];
  const float* bv = (const float*)d_in[6];
  const float* Wr = (const float*)d_in[7];
  const float* br = (const float*)d_in[8];
  float* out = (float*)d_out;

  char* ws = (char*)d_ws;
  __bf16* wqb    = (__bf16*)(ws + 0);         // 64*512*2
  __bf16* wkb    = (__bf16*)(ws + 65536);
  __bf16* wvb    = (__bf16*)(ws + 131072);
  __bf16* wrb    = (__bf16*)(ws + 196608);
  float*  scores = (float*) (ws + 262144);    // 8*64*64*4
  __bf16* wgt    = (__bf16*)(ws + 393216);    // 8*64*64*2
  __bf16* vws    = (__bf16*)(ws + 458752);    // 8*64*16384*2 = 16 MB

  k_prep<<<128, 256, 0, stream>>>(Wq, Wk, Wv, Wr, wqb, wkb, wvb, wrb, scores);
  k_qkv_scores<<<Bn * (Nn / NT), 256, 0, stream>>>(x, wqb, wkb, wvb,
                                                   bq, bk, bv, vws, scores);
  k_softmax<<<Bn, ICc, 0, stream>>>(scores, wgt);
  k_att_out<<<Bn * (Nn / NT), 256, 0, stream>>>(x, wgt, vws, wrb, br, out);
}
